// HGCN_45260365365980
// MI455X (gfx1250) — compile-verified, matrix operand order checked by reading, not run on hardware
//
#include <hip/hip_runtime.h>
#include <hip/hip_bf16.h>

// ---------------------------------------------------------------------------
// Types for CDNA5 WMMA (gfx1250, wave32)
// ---------------------------------------------------------------------------
typedef __bf16 bf16;
typedef __attribute__((ext_vector_type(16))) __bf16 v16bf;
typedef __attribute__((ext_vector_type(8)))  float  v8f;

static constexpr int kNodes   = 50000;
static constexpr int kInc     = 300000;
static constexpr int kHE      = 10000;
static constexpr int kGraphs  = 64;
static constexpr int kInC     = 768;
static constexpr int kHid     = 512;

// LDS A-tile row pitch in floats: 32 data + 4 pad -> lane stride 36 banks,
// gcd(36,64)=4 -> 16 lanes of a b128 read hit disjoint bank quads.
static constexpr int kApitch  = 36;

// ---------------------------------------------------------------------------
// A fragment (16x32 bf16, M x K) read from the LDS-staged f32 tile, converted
// to bf16 in-register. Lane L<16: row M=L, K={0..7, 16..23}; lanes 16..31:
// row M=L-16, K={8..15, 24..31}  (ISA 05_wmma 16-bit A layout).
// ---------------------------------------------------------------------------
__device__ __forceinline__ v16bf load_a_frag_lds(const float* As, int mrel, int lane) {
  const float* p = As + (mrel + (lane & 15)) * kApitch + ((lane & 16) ? 8 : 0);
  v16bf r;
#pragma unroll
  for (int j = 0; j < 8; ++j) {
    r[j]     = (bf16)p[j];
    r[j + 8] = (bf16)p[j + 16];
  }
  return r;
}

// B fragment (32x16 bf16, K x N): lane holds one column n, same K striping.
// Bt is the transposed weight [N][K] bf16 (<=768KB, L2-resident): two 16B loads.
__device__ __forceinline__ v16bf load_b_frag(const bf16* __restrict__ Bt, int K,
                                             int nbase, int kbase, int lane) {
  int col = nbase + (lane & 15);
  const bf16* p = Bt + (size_t)col * K + kbase + ((lane & 16) ? 8 : 0);
  v16bf r;
#pragma unroll
  for (int j = 0; j < 8; ++j) {
    r[j]     = p[j];
    r[j + 8] = p[j + 16];
  }
  return r;
}

// C/D layout: VGPR r -> (M = mb16 + r (+8 for lanes 16..31), N = nb16 + lane%16)
__device__ __forceinline__ void store_acc(float* __restrict__ C, const v8f& acc,
                                          int M, int N, int mb16, int nb16,
                                          int lane, const float* bias, int do_relu) {
  int n = nb16 + (lane & 15);
  float bv = bias ? bias[n] : 0.0f;
  int m0 = mb16 + ((lane & 16) ? 8 : 0);
#pragma unroll
  for (int r = 0; r < 8; ++r) {
    int m = m0 + r;
    if (m < M) {
      float v = acc[r] + bv;
      if (do_relu) v = fmaxf(v, 0.0f);
      C[(size_t)m * N + n] = v;
    }
  }
}

// ---------------------------------------------------------------------------
// Tiled WMMA GEMM: C[M,N] = act(A_f32[M,K] @ Bt_bf16[N,K]^T + bias)
// 256 threads = 8 waves; wave w -> 32x32 patch (2x2 WMMA tiles); block tile
// 128(M) x 64(N). A staged through a double-buffered LDS tile filled with
// GLOBAL_LOAD_ASYNC_TO_LDS_B128 (ASYNCcnt), overlapping the copy of K-step
// i+1 with the WMMAs of step i. K must be a multiple of 32 (768/512 here).
// ---------------------------------------------------------------------------
__global__ __launch_bounds__(256) void hgcn_wmma_gemm(
    const float* __restrict__ A, const bf16* __restrict__ Bt,
    const float* __restrict__ bias, float* __restrict__ C,
    int M, int N, int K, int do_relu) {
  __shared__ alignas(16) float As[2][128 * kApitch];   // 2 x 18KB

  const int tid    = threadIdx.x;
  const int lane   = tid & 31;
  const int w      = tid >> 5;                 // 0..7
  const int mtile  = (w & 3) * 32;             // 0,32,64,96 (relative to block)
  const int nb     = blockIdx.x * 64 + (w >> 2) * 32;
  const int blockM = blockIdx.y * 128;

  // Async-stage one 128x32 f32 tile: 1024 x 16B chunks, 4 per thread.
  auto issue_copy = [&](int buf, int kb) {
#pragma unroll
    for (int r = 0; r < 4; ++r) {
      int chunk = (r << 8) + tid;              // 0..1023
      int row   = chunk >> 3;                  // 0..127
      int c4    = (chunk & 7) << 2;            // 0,4,..,28 (floats)
      int grow  = blockM + row;
      if (grow >= M) grow = M - 1;             // clamp; stores are guarded
      const float* gp = A + (size_t)grow * K + kb + c4;
      unsigned lds = (unsigned)(uintptr_t)&As[buf][row * kApitch + c4];
      asm volatile("global_load_async_to_lds_b128 %0, %1, off"
                   :: "v"(lds), "v"(gp) : "memory");
    }
  };

  v8f acc00 = {}, acc01 = {}, acc10 = {}, acc11 = {};

  const int nk = K >> 5;
  issue_copy(0, 0);                            // prologue: buffer 0
  for (int i = 0; i < nk; ++i) {
    const int  cur  = i & 1;
    const bool more = (i + 1) < nk;
    __syncthreads();                           // buf !cur free to overwrite
    if (more) {
      issue_copy(cur ^ 1, (i + 1) << 5);
      // my 4 'cur' copies retired (in-order); 4 'next' may stay in flight
      asm volatile("s_wait_asynccnt 0x4" ::: "memory");
    } else {
      asm volatile("s_wait_asynccnt 0x0" ::: "memory");
    }
    __syncthreads();                           // all waves' 'cur' slices landed

    const int kb = i << 5;
    v16bf a0 = load_a_frag_lds(&As[cur][0], mtile,      lane);
    v16bf a1 = load_a_frag_lds(&As[cur][0], mtile + 16, lane);
    v16bf b0 = load_b_frag(Bt, K, nb,      kb, lane);
    v16bf b1 = load_b_frag(Bt, K, nb + 16, kb, lane);
    acc00 = __builtin_amdgcn_wmma_f32_16x16x32_bf16(false, a0, false, b0, (short)0, acc00, false, false);
    acc01 = __builtin_amdgcn_wmma_f32_16x16x32_bf16(false, a0, false, b1, (short)0, acc01, false, false);
    acc10 = __builtin_amdgcn_wmma_f32_16x16x32_bf16(false, a1, false, b0, (short)0, acc10, false, false);
    acc11 = __builtin_amdgcn_wmma_f32_16x16x32_bf16(false, a1, false, b1, (short)0, acc11, false, false);
  }

  const int mb = blockM + mtile;
  store_acc(C, acc00, M, N, mb,      nb,      lane, bias, do_relu);
  store_acc(C, acc01, M, N, mb,      nb + 16, lane, bias, do_relu);
  store_acc(C, acc10, M, N, mb + 16, nb,      lane, bias, do_relu);
  store_acc(C, acc11, M, N, mb + 16, nb + 16, lane, bias, do_relu);
}

// ---------------------------------------------------------------------------
// Helpers: zero, weight cast+transpose, degrees
// ---------------------------------------------------------------------------
__global__ void hgcn_zero(float* p, int n) {
  int i = blockIdx.x * blockDim.x + threadIdx.x;
  if (i < n) p[i] = 0.0f;
}

// W[K,N] f32 -> Bt[N,K] bf16
__global__ void hgcn_cast_transpose(const float* __restrict__ W,
                                    bf16* __restrict__ Bt, int K, int N) {
  int i = blockIdx.x * blockDim.x + threadIdx.x;
  if (i >= K * N) return;
  int n = i / K, k = i % K;
  Bt[i] = (bf16)W[(size_t)k * N + n];
}

__global__ void hgcn_degree(const int* __restrict__ ni, const int* __restrict__ hi,
                            float* Dc, float* Bc, int n) {
  int i = blockIdx.x * blockDim.x + threadIdx.x;
  if (i < n) {
    atomicAdd(&Dc[ni[i]], 1.0f);
    atomicAdd(&Bc[hi[i]], 1.0f);
  }
}

__global__ void hgcn_invert(float* p, int n) {
  int i = blockIdx.x * blockDim.x + threadIdx.x;
  if (i < n) { float v = p[i]; p[i] = v > 0.0f ? 1.0f / v : 0.0f; }
}

// ---------------------------------------------------------------------------
// Scatter phases (memory-bound; ~1.3 GB/layer -> ~60us at 23.3 TB/s)
// one block per incidence entry, coalesced 512-wide rows
// ---------------------------------------------------------------------------
__global__ __launch_bounds__(256) void hgcn_scatter_n2e(
    const float* __restrict__ t, const int* __restrict__ ni,
    const int* __restrict__ hi, float* e) {
  int i = blockIdx.x;
  int node = ni[i], he = hi[i];
  size_t sb = (size_t)node * kHid, db = (size_t)he * kHid;
  int c = threadIdx.x;
  atomicAdd(&e[db + c],       t[sb + c]);
  atomicAdd(&e[db + c + 256], t[sb + c + 256]);
}

__global__ __launch_bounds__(256) void hgcn_scatter_e2n(
    const float* __restrict__ e, const float* __restrict__ Binv,
    const int* __restrict__ ni, const int* __restrict__ hi, float* t2) {
  int i = blockIdx.x;
  int node = ni[i], he = hi[i];
  float bw = Binv[he];
  size_t sb = (size_t)he * kHid, db = (size_t)node * kHid;
  int c = threadIdx.x;
  atomicAdd(&t2[db + c],       e[sb + c] * bw);
  atomicAdd(&t2[db + c + 256], e[sb + c + 256] * bw);
}

// ---------------------------------------------------------------------------
// Post: y = relu(LayerNorm(t*Dinv + conv_b)) [+ residual] [+ h0]  -> h
// one block per node row (512 cols, 256 threads x 2), LDS tree reductions
// ---------------------------------------------------------------------------
__global__ __launch_bounds__(256) void hgcn_post(
    const float* __restrict__ t, const float* __restrict__ Dinv,
    const float* __restrict__ cb, const float* __restrict__ lg,
    const float* __restrict__ lb, const float* res, const float* h0,
    float* hout) {
  __shared__ float red[256];
  int row = blockIdx.x;
  int tid = threadIdx.x;
  float dinv = Dinv[row];
  size_t base = (size_t)row * kHid;
  float v0 = t[base + tid]       * dinv + cb[tid];
  float v1 = t[base + tid + 256] * dinv + cb[tid + 256];

  red[tid] = v0 + v1;
  __syncthreads();
  for (int s = 128; s > 0; s >>= 1) { if (tid < s) red[tid] += red[tid + s]; __syncthreads(); }
  float mean = red[0] * (1.0f / 512.0f);
  __syncthreads();
  red[tid] = v0 * v0 + v1 * v1;
  __syncthreads();
  for (int s = 128; s > 0; s >>= 1) { if (tid < s) red[tid] += red[tid + s]; __syncthreads(); }
  float var = red[0] * (1.0f / 512.0f) - mean * mean;
  float inv = rsqrtf(var + 1e-5f);

  float y0 = fmaxf((v0 - mean) * inv * lg[tid]       + lb[tid],       0.0f);
  float y1 = fmaxf((v1 - mean) * inv * lg[tid + 256] + lb[tid + 256], 0.0f);
  if (res) { y0 += res[base + tid]; y1 += res[base + tid + 256]; }   // res may alias hout (read-before-write per thread)
  if (h0)  { y0 += h0[base + tid];  y1 += h0[base + tid + 256]; }
  hout[base + tid]       = y0;
  hout[base + tid + 256] = y1;
}

// ---------------------------------------------------------------------------
// Pooling: per-graph sum + monotonic-uint-encoded float max
// ---------------------------------------------------------------------------
__global__ __launch_bounds__(256) void hgcn_pool(
    const float* __restrict__ h, const int* __restrict__ batch,
    float* gsum, unsigned* gmax, float* cnt) {
  int node = blockIdx.x;
  int b = batch[node];
  int c = threadIdx.x;
#pragma unroll
  for (int k = 0; k < 2; ++k) {
    int col = c + k * 256;
    float v = h[(size_t)node * kHid + col];
    atomicAdd(&gsum[b * kHid + col], v);
    unsigned u = __float_as_uint(v);
    unsigned enc = (u & 0x80000000u) ? ~u : (u | 0x80000000u);  // order-preserving
    atomicMax(&gmax[b * kHid + col], enc);
  }
  if (c == 0) atomicAdd(&cnt[b], 1.0f);
}

__global__ void hgcn_pool_fin(const float* __restrict__ gsum,
                              const unsigned* __restrict__ gmax,
                              const float* __restrict__ cnt, float* g) {
  int i = blockIdx.x * blockDim.x + threadIdx.x;
  if (i >= kGraphs * kHid) return;
  int b = i / kHid, c = i % kHid;
  float m = gsum[i] / fmaxf(cnt[b], 1.0f);
  unsigned e = gmax[i];
  unsigned u = (e & 0x80000000u) ? (e ^ 0x80000000u) : ~e;
  float mx = __uint_as_float(u);
  if (!isfinite(mx)) mx = 0.0f;
  g[(size_t)b * (2 * kHid) + c]        = m;
  g[(size_t)b * (2 * kHid) + kHid + c] = mx;
}

// ---------------------------------------------------------------------------
// Tiny MLP head (~88 MFLOP total): plain f32 per-output-element kernel
// ---------------------------------------------------------------------------
__global__ void hgcn_dense(const float* __restrict__ A, const float* __restrict__ W,
                           const float* __restrict__ b, float* __restrict__ out,
                           int M, int K, int N, int do_relu) {
  int i = blockIdx.x * blockDim.x + threadIdx.x;
  if (i >= M * N) return;
  int m = i / N, n = i % N;
  float acc = b[n];
  for (int k = 0; k < K; ++k) acc += A[(size_t)m * K + k] * W[(size_t)k * N + n];
  if (do_relu) acc = fmaxf(acc, 0.0f);
  out[(size_t)m * N + n] = acc;
}

// ---------------------------------------------------------------------------
// Orchestration
// ---------------------------------------------------------------------------
extern "C" void kernel_launch(void* const* d_in, const int* in_sizes, int n_in,
                              void* d_out, int out_size, void* d_ws, size_t ws_size,
                              hipStream_t stream) {
  const float* x       = (const float*)d_in[0];
  const int* node_idx  = (const int*)d_in[1];
  const int* hedge_idx = (const int*)d_in[2];
  const int* batch     = (const int*)d_in[3];
  const float* W_in    = (const float*)d_in[4];
  const float* b_in    = (const float*)d_in[5];
  const float* conv_W  = (const float*)d_in[6];
  const float* conv_b  = (const float*)d_in[7];
  const float* ln_g    = (const float*)d_in[8];
  const float* ln_b    = (const float*)d_in[9];
  const float* W_p0    = (const float*)d_in[10];
  const float* b_p0    = (const float*)d_in[11];
  const float* W_p1    = (const float*)d_in[12];
  const float* b_p1    = (const float*)d_in[13];
  const float* W_c0    = (const float*)d_in[14];
  const float* b_c0    = (const float*)d_in[15];
  const float* W_c1    = (const float*)d_in[16];
  const float* b_c1    = (const float*)d_in[17];

  char* ws = (char*)d_ws;
  size_t off = 0;
  auto alloc = [&](size_t bytes) {
    void* p = ws + off;
    off = (off + bytes + 255) & ~(size_t)255;
    return p;
  };

  float* h     = (float*)alloc((size_t)kNodes * kHid * 4);   // 102.4 MB
  float* h0    = (float*)alloc((size_t)kNodes * kHid * 4);   // 102.4 MB
  float* t     = (float*)alloc((size_t)kNodes * kHid * 4);   // 102.4 MB (gather src, reused as scatter dst)
  float* e     = (float*)alloc((size_t)kHE * kHid * 4);      // 20.5 MB
  bf16* WinT   = (bf16*)alloc((size_t)kHid * kInC * 2);      // 0.79 MB
  bf16* convT  = (bf16*)alloc((size_t)3 * kHid * kHid * 2);  // 1.57 MB
  float* Dinv  = (float*)alloc((size_t)kNodes * 4);
  float* Binv  = (float*)alloc((size_t)kHE * 4);
  float* gsum  = (float*)alloc((size_t)kGraphs * kHid * 4);
  unsigned* gmax = (unsigned*)alloc((size_t)kGraphs * kHid * 4);
  float* cnt   = (float*)alloc((size_t)kGraphs * 4);
  float* g     = (float*)alloc((size_t)kGraphs * 2 * kHid * 4);
  float* gp    = (float*)alloc((size_t)kGraphs * kHid * 4);
  float* gq    = (float*)alloc((size_t)kGraphs * 256 * 4);
  float* gr    = (float*)alloc((size_t)kGraphs * 128 * 4);

  auto cdiv = [](int a, int b) { return (a + b - 1) / b; };

  // --- weight cast + transpose (once per call; tiny) ---
  hgcn_cast_transpose<<<cdiv(kInC * kHid, 256), 256, 0, stream>>>(W_in, WinT, kInC, kHid);
  for (int l = 0; l < 3; ++l)
    hgcn_cast_transpose<<<cdiv(kHid * kHid, 256), 256, 0, stream>>>(
        conv_W + (size_t)l * kHid * kHid, convT + (size_t)l * kHid * kHid, kHid, kHid);

  // --- degree normalizers ---
  hgcn_zero<<<cdiv(kNodes, 256), 256, 0, stream>>>(Dinv, kNodes);
  hgcn_zero<<<cdiv(kHE, 256), 256, 0, stream>>>(Binv, kHE);
  hgcn_degree<<<cdiv(kInc, 256), 256, 0, stream>>>(node_idx, hedge_idx, Dinv, Binv, kInc);
  hgcn_invert<<<cdiv(kNodes, 256), 256, 0, stream>>>(Dinv, kNodes);
  hgcn_invert<<<cdiv(kHE, 256), 256, 0, stream>>>(Binv, kHE);

  // --- input projection: h = relu(x @ W_in + b_in)  (WMMA bf16) ---
  {
    dim3 grid(kHid / 64, cdiv(kNodes, 128));
    hgcn_wmma_gemm<<<grid, 256, 0, stream>>>(x, WinT, b_in, h, kNodes, kHid, kInC, 1);
  }
  hipMemcpyAsync(h0, h, (size_t)kNodes * kHid * 4, hipMemcpyDeviceToDevice, stream);

  // --- 3 hypergraph conv layers ---
  for (int l = 0; l < 3; ++l) {
    dim3 grid(kHid / 64, cdiv(kNodes, 128));
    hgcn_wmma_gemm<<<grid, 256, 0, stream>>>(
        h, convT + (size_t)l * kHid * kHid, nullptr, t, kNodes, kHid, kHid, 0);

    hgcn_zero<<<cdiv(kHE * kHid, 256), 256, 0, stream>>>(e, kHE * kHid);
    hgcn_scatter_n2e<<<kInc, 256, 0, stream>>>(t, node_idx, hedge_idx, e);

    hgcn_zero<<<cdiv(kNodes * kHid, 256), 256, 0, stream>>>(t, kNodes * kHid);
    hgcn_scatter_e2n<<<kInc, 256, 0, stream>>>(e, Binv, node_idx, hedge_idx, t);

    hgcn_post<<<kNodes, 256, 0, stream>>>(
        t, Dinv, conv_b + l * kHid, ln_g + l * kHid, ln_b + l * kHid,
        (l > 0) ? h : nullptr, (l == 2) ? h0 : nullptr, h);
  }

  // --- pooling ---
  hgcn_zero<<<cdiv(kGraphs * kHid, 256), 256, 0, stream>>>(gsum, kGraphs * kHid);
  hgcn_zero<<<cdiv(kGraphs * kHid, 256), 256, 0, stream>>>((float*)gmax, kGraphs * kHid);
  hgcn_zero<<<1, 64, 0, stream>>>(cnt, kGraphs);
  hgcn_pool<<<kNodes, 256, 0, stream>>>(h, batch, gsum, gmax, cnt);
  hgcn_pool_fin<<<cdiv(kGraphs * kHid, 256), 256, 0, stream>>>(gsum, gmax, cnt, g);

  // --- MLP head (tiny) ---
  hgcn_dense<<<cdiv(kGraphs * kHid, 256), 256, 0, stream>>>(g,  W_p0, b_p0, gp, kGraphs, 2 * kHid, kHid, 1);
  hgcn_dense<<<cdiv(kGraphs * 256, 256), 256, 0, stream>>>(gp, W_p1, b_p1, gq, kGraphs, kHid, 256, 0);
  hgcn_dense<<<cdiv(kGraphs * 128, 256), 256, 0, stream>>>(gq, W_c0, b_c0, gr, kGraphs, 256, 128, 1);
  hgcn_dense<<<1, 128, 0, stream>>>(gr, W_c1, b_c1, (float*)d_out, kGraphs, 128, 2, 0);
}
